// SUPRA_44255343018912
// MI455X (gfx1250) — matrix operand order repaired and checked by reading, not compiled, not run on hardware
//
#include <hip/hip_runtime.h>

// ---------------------------------------------------------------------------
// MI455X (gfx1250) multimodal-GCN pipeline, bf16 WMMA (v_wmma_f32_16x16x32_bf16)
// Round 2: use native f32->bf16 conversion (hardware v_cvt path) instead of
// manual integer RNE rounding to cut VALU overhead in the encoder GEMMs.
// ---------------------------------------------------------------------------

typedef __attribute__((ext_vector_type(16))) __bf16 v16bf;
typedef __attribute__((ext_vector_type(8)))  float  v8f;

#define N_NODES  100000
#define N_EDGES  1600000
#define D_TEXT   768
#define D_VIS    512
#define D_EMB    128
#define N_CLASSES 40

static __device__ __forceinline__ __bf16 f2bf(float x) {
    return static_cast<__bf16>(x);   // RNE; lowers to hardware cvt on gfx1250
}

union FragBF {
    v16bf v;
    uint4 u[2];
    __bf16 h[16];
};

// ---------------------------------------------------------------------------
// Pack a [K, Ncols] f32 weight into WMMA B-fragment layout:
// fragment (kt, nt) = 32 lanes x 16 bf16, element e of lane l is
//   W[kt*32 + (l>=16 ? 8 : 0) + (e<8 ? e : e+8)][nt*16 + (l&15)]  (zero-padded past Ncols)
// stored at P[(kt*NT + nt)*512 + l*16 + e].
// ---------------------------------------------------------------------------
__global__ void pack_w_kernel(const float* __restrict__ W, __bf16* __restrict__ P,
                              int Ncols, int NT, int total) {
    int idx = blockIdx.x * blockDim.x + threadIdx.x;
    if (idx >= total) return;
    int f = idx >> 9, r = idx & 511;
    int lane = r >> 4, e = r & 15;
    int kt = f / NT, nt = f - kt * NT;
    int k = (kt << 5) + ((lane >= 16) ? 8 : 0) + (e < 8 ? e : e + 8);
    int n = (nt << 4) + (lane & 15);
    float v = (n < Ncols) ? W[(size_t)k * Ncols + n] : 0.0f;
    P[idx] = f2bf(v);
}

// ---------------------------------------------------------------------------
// WMMA GEMM: Out[nRows, ldo] = op(A[nRows, lda-strided] @ Wpacked (+bias))
// One wave computes a 16-row x (NT*16)-col strip; K in steps of 32.
// ---------------------------------------------------------------------------
template <int NT, bool A_F32, bool RELU, bool BIAS, bool OUT_BF16>
__global__ __launch_bounds__(256) void gemm_wmma_kernel(
    const void* __restrict__ Aopq, const __bf16* __restrict__ Bp,
    const float* __restrict__ bias, void* __restrict__ Out,
    int nRows, int lda, int K, int ldo, int ncols) {

    int lane = threadIdx.x & 31;
    int wave = (blockIdx.x * blockDim.x + threadIdx.x) >> 5;
    int m0 = wave << 4;
    if (m0 >= nRows) return;                      // wave-uniform: EXEC stays full

    const int KT = K >> 5;
    const int mrow  = m0 + (lane & 15);
    const int khalf = (lane >> 4) << 3;           // 0 or 8

    v8f zero = {};
    v8f acc[NT];
#pragma unroll
    for (int i = 0; i < NT; ++i) acc[i] = zero;

    for (int kt = 0; kt < KT; ++kt) {
        const int kb = (kt << 5) + khalf;
        FragBF a;
        if (A_F32) {
            const float* Ar = (const float*)Aopq + (size_t)mrow * lda + kb;
            float4 f0 = *(const float4*)(Ar);
            float4 f1 = *(const float4*)(Ar + 4);
            float4 f2 = *(const float4*)(Ar + 16);
            float4 f3 = *(const float4*)(Ar + 20);
            a.h[0]  = f2bf(f0.x); a.h[1]  = f2bf(f0.y); a.h[2]  = f2bf(f0.z); a.h[3]  = f2bf(f0.w);
            a.h[4]  = f2bf(f1.x); a.h[5]  = f2bf(f1.y); a.h[6]  = f2bf(f1.z); a.h[7]  = f2bf(f1.w);
            a.h[8]  = f2bf(f2.x); a.h[9]  = f2bf(f2.y); a.h[10] = f2bf(f2.z); a.h[11] = f2bf(f2.w);
            a.h[12] = f2bf(f3.x); a.h[13] = f2bf(f3.y); a.h[14] = f2bf(f3.z); a.h[15] = f2bf(f3.w);
        } else {
            const __bf16* Ar = (const __bf16*)Aopq + (size_t)mrow * lda + kb;
            a.u[0] = *(const uint4*)(Ar);
            a.u[1] = *(const uint4*)(Ar + 16);
        }
        const __bf16* bbase = Bp + (((size_t)kt * NT) << 9) + (lane << 4);
#pragma unroll
        for (int nt = 0; nt < NT; ++nt) {
            FragBF b;
            b.u[0] = *(const uint4*)(bbase + ((size_t)nt << 9));
            b.u[1] = *(const uint4*)(bbase + ((size_t)nt << 9) + 8);
            acc[nt] = __builtin_amdgcn_wmma_f32_16x16x32_bf16(
                false, a.v, false, b.v, (short)0, acc[nt], false, false);
        }
    }

    const int rbase = m0 + khalf;                 // C layout: lane<16 -> M=i, else M=8+i
#pragma unroll
    for (int nt = 0; nt < NT; ++nt) {
        int col = (nt << 4) + (lane & 15);
        if (col < ncols) {
            float bv = BIAS ? bias[col] : 0.0f;
#pragma unroll
            for (int i = 0; i < 8; ++i) {
                float v = acc[nt][i] + bv;
                if (RELU) v = fmaxf(v, 0.0f);
                size_t o = (size_t)(rbase + i) * ldo + col;
                if (OUT_BF16) ((__bf16*)Out)[o] = f2bf(v);
                else          ((float*)Out)[o]  = v;
            }
        }
    }
}

// ---------------------------------------------------------------------------
// Degrees, norms, scatter-aggregate, bias+convert, final combine
// ---------------------------------------------------------------------------
__global__ void deg_kernel(const int* __restrict__ src, const int* __restrict__ dst,
                           float* __restrict__ dout, float* __restrict__ din, int E) {
    int e = blockIdx.x * blockDim.x + threadIdx.x;
    if (e >= E) return;
    atomicAdd(&dout[src[e]], 1.0f);
    atomicAdd(&din[dst[e]], 1.0f);
}

__global__ void invrsqrt_kernel(float* __restrict__ a, int n) {
    int i = blockIdx.x * blockDim.x + threadIdx.x;
    if (i >= n) return;
    a[i] = rsqrtf(fmaxf(a[i], 1.0f));
}

// one wave per edge: lane covers 4 of the 128 features
__global__ __launch_bounds__(256) void scatter_kernel(
    const float* __restrict__ t, const int* __restrict__ src,
    const int* __restrict__ dst, const float* __restrict__ iout,
    const float* __restrict__ iin, float* __restrict__ agg, int E) {
    int lane = threadIdx.x & 31;
    int e = (blockIdx.x << 3) + (threadIdx.x >> 5);
    if (e >= E) return;
    int s = src[e], d = dst[e];
    float nrm = iout[s] * iin[d];
    float4 v = *(const float4*)(t + ((size_t)s << 7) + (lane << 2));
    float* ap = agg + ((size_t)d << 7) + (lane << 2);
    atomicAdd(ap + 0, v.x * nrm);
    atomicAdd(ap + 1, v.y * nrm);
    atomicAdd(ap + 2, v.z * nrm);
    atomicAdd(ap + 3, v.w * nrm);
}

__global__ void biasconv_kernel(const float* __restrict__ agg, const float* __restrict__ b,
                                __bf16* __restrict__ h, int total) {
    int i = blockIdx.x * blockDim.x + threadIdx.x;
    if (i >= total) return;
    h[i] = f2bf(agg[i] + b[i & 127]);
}

__global__ void combine_kernel(const float* __restrict__ C, const float* __restrict__ Ut,
                               const float* __restrict__ Uv, float* __restrict__ F, int total) {
    int i = blockIdx.x * blockDim.x + threadIdx.x;
    if (i >= total) return;
    F[i] = (C[i] + Ut[i] + Uv[i]) * (1.0f / 3.0f);
}

// ---------------------------------------------------------------------------
// Host launcher
// ---------------------------------------------------------------------------
extern "C" void kernel_launch(void* const* d_in, const int* in_sizes, int n_in,
                              void* d_out, int out_size, void* d_ws, size_t ws_size,
                              hipStream_t stream) {
    (void)in_sizes; (void)n_in; (void)out_size; (void)ws_size;

    const float* text = (const float*)d_in[0];
    const float* vis  = (const float*)d_in[1];
    const int*   esrc = (const int*)d_in[2];
    const int*   edst = (const int*)d_in[3];
    const float* W_t  = (const float*)d_in[4];
    const float* b_t  = (const float*)d_in[5];
    const float* W_v  = (const float*)d_in[6];
    const float* b_v  = (const float*)d_in[7];
    const float* W_m0 = (const float*)d_in[8];
    const float* b_m0 = (const float*)d_in[9];
    const float* W_m1 = (const float*)d_in[10];
    const float* b_m1 = (const float*)d_in[11];
    const float* W_c  = (const float*)d_in[12];
    const float* b_c  = (const float*)d_in[13];
    const float* W_ut = (const float*)d_in[14];
    const float* b_ut = (const float*)d_in[15];
    const float* W_uv = (const float*)d_in[16];
    const float* b_uv = (const float*)d_in[17];

    // ---- workspace layout ----
    char* ws = (char*)d_ws;
    size_t off = 0;
    auto take = [&](size_t bytes) { char* p = ws + off; off += (bytes + 255) & ~(size_t)255; return p; };

    __bf16* pWt  = (__bf16*)take((size_t)24 * 8 * 512 * 2);   // 768x128
    __bf16* pWv  = (__bf16*)take((size_t)16 * 8 * 512 * 2);   // 512x128
    __bf16* pW0  = (__bf16*)take((size_t)8  * 8 * 512 * 2);   // 256x128
    __bf16* pW1  = (__bf16*)take((size_t)4  * 8 * 512 * 2);   // 128x128
    __bf16* pWc  = (__bf16*)take((size_t)4  * 3 * 512 * 2);   // 128x40 (pad 48)
    __bf16* pWut = (__bf16*)take((size_t)4  * 3 * 512 * 2);
    __bf16* pWuv = (__bf16*)take((size_t)4  * 3 * 512 * 2);
    __bf16* c_bf = (__bf16*)take((size_t)N_NODES * 256 * 2);  // concat(e_t,e_v) bf16
    float*  tbuf = (float*)take((size_t)N_NODES * 128 * 4);   // GEMM output before scatter
    float*  agg  = (float*)take((size_t)N_NODES * 128 * 4);   // scatter accumulator
    __bf16* h0   = (__bf16*)take((size_t)N_NODES * 128 * 2);
    __bf16* h1   = (__bf16*)take((size_t)N_NODES * 128 * 2);
    float*  dego = (float*)take((size_t)N_NODES * 4);
    float*  degi = (float*)take((size_t)N_NODES * 4);

    float* out      = (float*)d_out;
    const int OUTN  = N_NODES * N_CLASSES;
    float* outC  = out + (size_t)OUTN;
    float* outUt = out + (size_t)2 * OUTN;
    float* outUv = out + (size_t)3 * OUTN;

    // ---- degrees (zero each call for determinism) ----
    hipMemsetAsync(dego, 0, (size_t)N_NODES * 4, stream);
    hipMemsetAsync(degi, 0, (size_t)N_NODES * 4, stream);

    // ---- pack weights into WMMA fragment layout ----
    auto packs = [&](const float* W, __bf16* P, int Ncols, int KT, int NT) {
        int total = KT * NT * 512;
        pack_w_kernel<<<(total + 255) / 256, 256, 0, stream>>>(W, P, Ncols, NT, total);
    };
    packs(W_t,  pWt,  128, 24, 8);
    packs(W_v,  pWv,  128, 16, 8);
    packs(W_m0, pW0,  128,  8, 8);
    packs(W_m1, pW1,  128,  4, 8);
    packs(W_c,  pWc,   40,  4, 3);
    packs(W_ut, pWut,  40,  4, 3);
    packs(W_uv, pWuv,  40,  4, 3);

    deg_kernel<<<(N_EDGES + 255) / 256, 256, 0, stream>>>(esrc, edst, dego, degi, N_EDGES);
    invrsqrt_kernel<<<(N_NODES + 255) / 256, 256, 0, stream>>>(dego, N_NODES);
    invrsqrt_kernel<<<(N_NODES + 255) / 256, 256, 0, stream>>>(degi, N_NODES);

    const int gemm_blocks = ((N_NODES / 16) + 7) / 8;   // 8 waves/block, 16 rows/wave

    // ---- modality encoders: relu(X @ W + b) -> bf16 concat buffer ----
    gemm_wmma_kernel<8, true, true, true, true><<<gemm_blocks, 256, 0, stream>>>(
        text, pWt, b_t, c_bf, N_NODES, D_TEXT, D_TEXT, 256, 128);
    gemm_wmma_kernel<8, true, true, true, true><<<gemm_blocks, 256, 0, stream>>>(
        vis, pWv, b_v, c_bf + 128, N_NODES, D_VIS, D_VIS, 256, 128);

    // ---- GCN layer 0: t = c @ W_mp0; agg = scatter; h0 = bf16(agg + b_mp0) ----
    gemm_wmma_kernel<8, false, false, false, false><<<gemm_blocks, 256, 0, stream>>>(
        c_bf, pW0, nullptr, tbuf, N_NODES, 256, 256, 128, 128);
    hipMemsetAsync(agg, 0, (size_t)N_NODES * 128 * 4, stream);
    scatter_kernel<<<(N_EDGES + 7) / 8, 256, 0, stream>>>(tbuf, esrc, edst, dego, degi, agg, N_EDGES);
    biasconv_kernel<<<(N_NODES * 128 + 255) / 256, 256, 0, stream>>>(agg, b_m0, h0, N_NODES * 128);

    // ---- GCN layer 1 ----
    gemm_wmma_kernel<8, false, false, false, false><<<gemm_blocks, 256, 0, stream>>>(
        h0, pW1, nullptr, tbuf, N_NODES, 128, 128, 128, 128);
    hipMemsetAsync(agg, 0, (size_t)N_NODES * 128 * 4, stream);
    scatter_kernel<<<(N_EDGES + 7) / 8, 256, 0, stream>>>(tbuf, esrc, edst, dego, degi, agg, N_EDGES);
    biasconv_kernel<<<(N_NODES * 128 + 255) / 256, 256, 0, stream>>>(agg, b_m1, h1, N_NODES * 128);

    // ---- classification heads (N=40, padded to 48 = 3 n-tiles) ----
    gemm_wmma_kernel<3, false, false, true, false><<<gemm_blocks, 256, 0, stream>>>(
        h1, pWc, b_c, outC, N_NODES, 128, 128, N_CLASSES, N_CLASSES);
    gemm_wmma_kernel<3, false, false, true, false><<<gemm_blocks, 256, 0, stream>>>(
        c_bf, pWut, b_ut, outUt, N_NODES, 256, 128, N_CLASSES, N_CLASSES);
    gemm_wmma_kernel<3, false, false, true, false><<<gemm_blocks, 256, 0, stream>>>(
        c_bf + 128, pWuv, b_uv, outUv, N_NODES, 256, 128, N_CLASSES, N_CLASSES);

    // ---- final average ----
    combine_kernel<<<(OUTN + 255) / 256, 256, 0, stream>>>(outC, outUt, outUv, out, OUTN);
}
